// model_7035156431102
// MI455X (gfx1250) — matrix operand order repaired and checked by reading, not compiled, not run on hardware
//
#include <hip/hip_runtime.h>
#include <hip/hip_bf16.h>
#include <math.h>

// ---------------- problem dims ----------------
#define B_    16
#define T_    128
#define NE_   64
#define H_    512
#define NH_   8
#define DH_   64
#define TGT_  16000
#define H2_   1024
#define H4_   2048
#define ROWS_ (B_ * T_)        // 2048
#define OUTC_ (TGT_ + NE_)     // 16064
#define RWG_  16               // workgroups in persistent recurrent kernel

typedef __attribute__((ext_vector_type(16))) __bf16 v16bf;
typedef __attribute__((ext_vector_type(8)))  float  v8f;

union FragBF { v16bf v; uint4 q[2]; };

__device__ __forceinline__ unsigned short f2bf(float f) {
  union { float f; unsigned u; } c; c.f = f;
  unsigned u = c.u;
  unsigned r = u + 0x7fffu + ((u >> 16) & 1u);   // round-to-nearest-even
  return (unsigned short)(r >> 16);
}
__device__ __forceinline__ float sigm(float x) { return 1.0f / (1.0f + __expf(-x)); }

// ---------------- prep kernels ----------------
__global__ void cvt_f32_bf16(const float* __restrict__ s, unsigned short* __restrict__ d, long long n) {
  long long i = (long long)blockIdx.x * blockDim.x + threadIdx.x;
  if (i < n) d[i] = f2bf(s[i]);
}

// U[n, 0:512] = W_ih[n, 0:512] (multiplies a), U[n, 512:1024] = W_hh[n, :] (multiplies hx)
__global__ void build_U(const float* __restrict__ W_ih, const float* __restrict__ W_hh,
                        const float* __restrict__ b_ih, const float* __restrict__ b_hh,
                        unsigned short* __restrict__ U, float* __restrict__ bc) {
  int i = blockIdx.x * blockDim.x + threadIdx.x;
  if (i < H4_ * H2_) {
    int row = i >> 10, col = i & (H2_ - 1);
    float v = (col < H_) ? W_ih[row * H2_ + col] : W_hh[row * H_ + (col - H_)];
    U[i] = f2bf(v);
  }
  if (i < H4_) bc[i] = b_ih[i] + b_hh[i];
}

// We[n, j] = W_ih[n, 512 + j]  (multiplies the embedding k_t)
__global__ void build_We(const float* __restrict__ W_ih, unsigned short* __restrict__ We) {
  int i = blockIdx.x * blockDim.x + threadIdx.x;
  if (i >= H4_ * H_) return;
  int row = i >> 9, col = i & (H_ - 1);
  We[i] = f2bf(W_ih[row * H2_ + H_ + col]);
}

__global__ void gather_emb(const int* __restrict__ outp, const float* __restrict__ emb,
                           unsigned short* __restrict__ e_bf) {
  int i = blockIdx.x * blockDim.x + threadIdx.x;
  if (i >= ROWS_ * H_) return;
  int row = i >> 9, col = i & (H_ - 1);
  e_bf[i] = f2bf(emb[(long long)outp[row] * H_ + col]);
}

// hx0 = mean over entities; cx = hx0; x = [a=0 | bf16(hx0)]; zero the global barrier
__global__ void init_state(const float* __restrict__ ents, float* __restrict__ cx,
                           unsigned short* __restrict__ x_bf, unsigned* __restrict__ bar) {
  int i = blockIdx.x * blockDim.x + threadIdx.x;
  if (i == 0) bar[0] = 0u;
  if (i >= B_ * H_) return;
  int b = i >> 9, j = i & (H_ - 1);
  float s = 0.f;
  for (int n = 0; n < NE_; ++n) s += ents[((long long)b * NE_ + n) * H_ + j];
  s *= (1.0f / NE_);
  cx[i] = s;
  x_bf[b * H2_ + j]      = 0;         // a0 = 0
  x_bf[b * H2_ + H_ + j] = f2bf(s);   // hx0
}

// ---------------- generic bf16 WMMA GEMM:  C[M,N] = A[M,K] @ B[N,K]^T (+bias) ----------------
// block = 256 threads = 8 waves; wave tile 32M x 64N (2 A frags x 4 N tiles = 8 accumulators);
// block tile 64M x 256N. Requirements: M % 64 == 0, N % 16 == 0, K % 32 == 0.
// No divergent control flow around WMMA: OOB N-tiles are computed on a clamped (valid)
// B row and only the stores are skipped (via a wave-uniform scalarized branch).
__global__ void __launch_bounds__(256)
gemm_wmma(const unsigned short* __restrict__ A, const unsigned short* __restrict__ B,
          float* __restrict__ C, unsigned short* __restrict__ Cbf, const float* __restrict__ bias,
          int M, int N, int K, int lda, int ldb, int ldc,
          long long sA, long long sB, long long sC) {
  A += (long long)blockIdx.z * sA;
  B += (long long)blockIdx.z * sB;
  long long cbase = (long long)blockIdx.z * sC;

  const int wave = threadIdx.x >> 5, lane = threadIdx.x & 31;
  const int wm = wave >> 2, wn = wave & 3;
  const int mbase  = blockIdx.y * 64 + wm * 32;
  const int nbase0 = blockIdx.x * 256 + wn * 64;
  const int lm  = lane & 15;
  const int khA = (lane >> 4) * 16;     // A frag: one 32B chunk at k+khA
  const int khB = (lane >> 4) * 16;     // B frag: one 32B chunk at k+khB

  int nb[4], nload[4];
#pragma unroll
  for (int j = 0; j < 4; ++j) {
    nb[j] = nbase0 + j * 16;
    nload[j] = (nb[j] < N) ? nb[j] : (N - 16);   // clamp OOB tiles to a valid B row
  }

  v8f acc[8] = {};   // acc[2j + i]: N-tile j, M-subtile i
  const unsigned short* pa0 = A + (long long)(mbase + lm) * lda + khA;
  const unsigned short* pa1 = pa0 + (long long)16 * lda;
  const unsigned short* pb0 = B + (long long)(nload[0] + lm) * ldb + khB;
  const unsigned short* pb1 = B + (long long)(nload[1] + lm) * ldb + khB;
  const unsigned short* pb2 = B + (long long)(nload[2] + lm) * ldb + khB;
  const unsigned short* pb3 = B + (long long)(nload[3] + lm) * ldb + khB;

  for (int kk = 0; kk < K; kk += 32) {
    FragBF a0, a1, b0, b1, b2, b3;
    a0.q[0] = *(const uint4*)(pa0 + kk);
    a0.q[1] = *(const uint4*)(pa0 + kk + 8);
    a1.q[0] = *(const uint4*)(pa1 + kk);
    a1.q[1] = *(const uint4*)(pa1 + kk + 8);
    b0.q[0] = ((const uint4*)(pb0 + kk))[0];  b0.q[1] = ((const uint4*)(pb0 + kk))[1];
    b1.q[0] = ((const uint4*)(pb1 + kk))[0];  b1.q[1] = ((const uint4*)(pb1 + kk))[1];
    b2.q[0] = ((const uint4*)(pb2 + kk))[0];  b2.q[1] = ((const uint4*)(pb2 + kk))[1];
    b3.q[0] = ((const uint4*)(pb3 + kk))[0];  b3.q[1] = ((const uint4*)(pb3 + kk))[1];
    acc[0] = __builtin_amdgcn_wmma_f32_16x16x32_bf16(false, a0.v, false, b0.v, (short)0, acc[0], false, false);
    acc[1] = __builtin_amdgcn_wmma_f32_16x16x32_bf16(false, a1.v, false, b0.v, (short)0, acc[1], false, false);
    acc[2] = __builtin_amdgcn_wmma_f32_16x16x32_bf16(false, a0.v, false, b1.v, (short)0, acc[2], false, false);
    acc[3] = __builtin_amdgcn_wmma_f32_16x16x32_bf16(false, a1.v, false, b1.v, (short)0, acc[3], false, false);
    acc[4] = __builtin_amdgcn_wmma_f32_16x16x32_bf16(false, a0.v, false, b2.v, (short)0, acc[4], false, false);
    acc[5] = __builtin_amdgcn_wmma_f32_16x16x32_bf16(false, a1.v, false, b2.v, (short)0, acc[5], false, false);
    acc[6] = __builtin_amdgcn_wmma_f32_16x16x32_bf16(false, a0.v, false, b3.v, (short)0, acc[6], false, false);
    acc[7] = __builtin_amdgcn_wmma_f32_16x16x32_bf16(false, a1.v, false, b3.v, (short)0, acc[7], false, false);
  }

#pragma unroll
  for (int j = 0; j < 4; ++j) {
    if (__builtin_amdgcn_readfirstlane(nb[j]) < N) {   // wave-uniform -> scalar branch
      int ncol = nb[j] + lm;
      float bi = bias ? bias[ncol] : 0.f;
#pragma unroll
      for (int i = 0; i < 2; ++i) {
#pragma unroll
        for (int r = 0; r < 8; ++r) {
          int m = mbase + i * 16 + r + ((lane >> 4) << 3);
          long long idx = cbase + (long long)m * ldc + ncol;
          float v = acc[2 * j + i][r] + bi;
          if (C)   C[idx] = v;
          if (Cbf) Cbf[idx] = f2bf(v);
        }
      }
    }
  }
}

// ---------------- persistent recurrent kernel (16 WGs x 256 threads = 128 waves) ----------------
__device__ __forceinline__ void gsync(unsigned* bar) {
  __syncthreads();
  if (threadIdx.x == 0) {
    __threadfence();
    unsigned my = __hip_atomic_fetch_add(bar, 1u, __ATOMIC_ACQ_REL, __HIP_MEMORY_SCOPE_AGENT);
    unsigned target = (my / RWG_ + 1u) * RWG_;
    while (__hip_atomic_load(bar, __ATOMIC_ACQUIRE, __HIP_MEMORY_SCOPE_AGENT) < target)
      __builtin_amdgcn_s_sleep(1);
  }
  __syncthreads();
}

__global__ void __launch_bounds__(256)
recurrent_kernel(unsigned short* __restrict__ x_bf,         // [16,1024]  [a|hx] bf16
                 const unsigned short* __restrict__ U_bf,   // [2048,1024]
                 const float* __restrict__ bc,              // [2048]
                 const float* __restrict__ Ge,              // [2048,2048] rows b*T+t
                 float* __restrict__ gbuf,                  // [16,2048]
                 float* __restrict__ cxbuf,                 // [16,512]
                 const unsigned short* __restrict__ Wq_bf,  // [512,512]
                 float* __restrict__ qbuf,                  // [16,512]
                 const float* __restrict__ Kbuf,            // [16,64,512]
                 const float* __restrict__ Vbuf,            // [16,64,512]
                 const int* __restrict__ entlens,           // [16]
                 float* __restrict__ lbuf,                  // [2048,1024]
                 unsigned short* __restrict__ l_bf,         // [2048,1024]
                 unsigned* __restrict__ bar) {
  __builtin_amdgcn_s_cluster_barrier();   // NOP when not launched as a cluster (ISA 6.6)

  const int lane = threadIdx.x & 31;
  const int wid  = threadIdx.x >> 5;
  const int wgid = blockIdx.x * 8 + wid;   // 0..127 (wave-uniform)
  const int khA = (lane >> 4) * 16;
  const int khB = (lane >> 4) * 16;
  const int rowA = lane & 15;

  __shared__ float satt[8][NE_];

  for (int t = 0; t < T_; ++t) {
    // ---- P1: gates = x @ U^T + Ge[:,t,:] + bc   (one 16-wide N tile per wave) ----
    {
      int nbase = wgid * 16;
      v8f acc = {};
      const unsigned short* pa = x_bf + rowA * H2_ + khA;
      const unsigned short* pb = U_bf + (long long)(nbase + rowA) * H2_ + khB;
      for (int kk = 0; kk < H2_; kk += 32) {
        FragBF a, b;
        a.q[0] = *(const uint4*)(pa + kk);
        a.q[1] = *(const uint4*)(pa + kk + 8);
        b.q[0] = ((const uint4*)(pb + kk))[0];
        b.q[1] = ((const uint4*)(pb + kk))[1];
        acc = __builtin_amdgcn_wmma_f32_16x16x32_bf16(false, a.v, false, b.v,
                                                      (short)0, acc, false, false);
      }
      int col = nbase + (lane & 15);
      float bi = bc[col];
#pragma unroll
      for (int r = 0; r < 8; ++r) {
        int m = r + ((lane >> 4) << 3);   // batch row 0..15
        gbuf[m * H4_ + col] = acc[r] + bi + Ge[((long long)(m * T_ + t)) * H4_ + col];
      }
    }
    gsync(bar);

    // ---- P2: LSTM pointwise ----
    for (int i = blockIdx.x * 256 + threadIdx.x; i < B_ * H_; i += RWG_ * 256) {
      int b = i >> 9, j = i & (H_ - 1);
      float ig = gbuf[b * H4_ + j];
      float fg = gbuf[b * H4_ + H_ + j];
      float cg = gbuf[b * H4_ + 2 * H_ + j];
      float og = gbuf[b * H4_ + 3 * H_ + j];
      float c2 = sigm(fg) * cxbuf[i] + sigm(ig) * tanhf(cg);
      float h2 = sigm(og) * tanhf(c2);
      cxbuf[i] = c2;
      x_bf[b * H2_ + H_ + j] = f2bf(h2);
      long long li = ((long long)(b * T_ + t)) * H2_ + j;
      lbuf[li] = h2;
      l_bf[li] = f2bf(h2);
    }
    gsync(bar);

    // ---- P3: q = hx2 @ Wq^T  (32 N tiles -> waves 0..31; scalarized branch keeps EXEC full) ----
    if (__builtin_amdgcn_readfirstlane(wgid) < 32) {
      int nbase = wgid * 16;
      v8f acc = {};
      const unsigned short* pa = x_bf + rowA * H2_ + H_ + khA;
      const unsigned short* pb = Wq_bf + (long long)(nbase + rowA) * H_ + khB;
      for (int kk = 0; kk < H_; kk += 32) {
        FragBF a, b;
        a.q[0] = *(const uint4*)(pa + kk);
        a.q[1] = *(const uint4*)(pa + kk + 8);
        b.q[0] = ((const uint4*)(pb + kk))[0];
        b.q[1] = ((const uint4*)(pb + kk))[1];
        acc = __builtin_amdgcn_wmma_f32_16x16x32_bf16(false, a.v, false, b.v,
                                                      (short)0, acc, false, false);
      }
      int col = nbase + (lane & 15);
#pragma unroll
      for (int r = 0; r < 8; ++r) {
        int m = r + ((lane >> 4) << 3);
        qbuf[m * H_ + col] = acc[r];
      }
    }
    gsync(bar);

    // ---- P4: attention, one (b, h) per wave ----
    {
      int b = wgid >> 3, h = wgid & 7;
      const float* qv = qbuf + b * H_ + h * DH_;
      int n0 = lane, n1 = lane + 32;
      const float* K0 = Kbuf + ((long long)(b * NE_ + n0)) * H_ + h * DH_;
      const float* K1 = Kbuf + ((long long)(b * NE_ + n1)) * H_ + h * DH_;
      float s0 = 0.f, s1 = 0.f;
      for (int d = 0; d < DH_; ++d) {
        float qd = qv[d];
        s0 += qd * K0[d];
        s1 += qd * K1[d];
      }
      s0 *= 0.125f; s1 *= 0.125f;   // 1/sqrt(64)
      int el = entlens[b];
      bool v0 = (n0 <= el), v1 = (n1 <= el);
      float mx = fmaxf(v0 ? s0 : -INFINITY, v1 ? s1 : -INFINITY);
      for (int off = 16; off > 0; off >>= 1) mx = fmaxf(mx, __shfl_xor(mx, off));
      float e0 = v0 ? __expf(s0 - mx) : 0.f;
      float e1 = v1 ? __expf(s1 - mx) : 0.f;
      float sm = e0 + e1;
      for (int off = 16; off > 0; off >>= 1) sm += __shfl_xor(sm, off);
      float inv = 1.0f / sm;
      satt[wid][n0] = e0 * inv;
      satt[wid][n1] = e1 * inv;
      __syncthreads();
      int d0 = lane, d1 = lane + 32;
      float a0 = 0.f, a1 = 0.f;
      for (int n = 0; n < NE_; ++n) {
        float at = satt[wid][n];
        const float* Vr = Vbuf + ((long long)(b * NE_ + n)) * H_ + h * DH_;
        a0 += at * Vr[d0];
        a1 += at * Vr[d1];
      }
      long long li = ((long long)(b * T_ + t)) * H2_ + H_ + h * DH_;
      lbuf[li + d0] = a0;  lbuf[li + d1] = a1;
      l_bf[li + d0] = f2bf(a0);  l_bf[li + d1] = f2bf(a1);
      x_bf[b * H2_ + h * DH_ + d0] = f2bf(a0);
      x_bf[b * H2_ + h * DH_ + d1] = f2bf(a1);
    }
    gsync(bar);
  }
}

// ---------------- epilogue kernels ----------------
__global__ void switch_kernel(const float* __restrict__ lbuf, const float* __restrict__ Wsw,
                              const float* __restrict__ bsw, float* __restrict__ ssw) {
  __shared__ float red[256];
  int r = blockIdx.x, tid = threadIdx.x;
  float p = 0.f;
  for (int c = tid; c < H2_; c += 256) p += lbuf[(long long)r * H2_ + c] * Wsw[c];
  red[tid] = p; __syncthreads();
  for (int s = 128; s > 0; s >>= 1) { if (tid < s) red[tid] += red[tid + s]; __syncthreads(); }
  if (tid == 0) ssw[r] = sigm(red[0] + bsw[0]);
}

__global__ void softmax_o_kernel(float* __restrict__ out1, const float* __restrict__ ssw) {
  __shared__ float red[256];
  int r = blockIdx.x, tid = threadIdx.x;
  float* lg = out1 + (long long)r * OUTC_;
  float m = -INFINITY;
  for (int c = tid; c < TGT_; c += 256) m = fmaxf(m, lg[c]);
  red[tid] = m; __syncthreads();
  for (int s = 128; s > 0; s >>= 1) { if (tid < s) red[tid] = fmaxf(red[tid], red[tid + s]); __syncthreads(); }
  m = red[0]; __syncthreads();
  float p = 0.f;
  for (int c = tid; c < TGT_; c += 256) p += __expf(lg[c] - m);
  red[tid] = p; __syncthreads();
  for (int s = 128; s > 0; s >>= 1) { if (tid < s) red[tid] += red[tid + s]; __syncthreads(); }
  float inv = ssw[r] / red[0];
  for (int c = tid; c < TGT_; c += 256) lg[c] = logf(__expf(lg[c] - m) * inv + 1e-6f);
}

__global__ void z_kernel(const float* __restrict__ unbuf, const int* __restrict__ entlens,
                         const float* __restrict__ ssw, float* __restrict__ out1,
                         float* __restrict__ out2) {
  __shared__ float red[64];
  int r = blockIdx.x, n = threadIdx.x;
  int b = r / T_;
  int el = entlens[b];
  bool valid = (n <= el);
  float u = unbuf[(long long)r * NE_ + n];
  float sc = valid ? u : -INFINITY;
  red[n] = sc; __syncthreads();
  for (int s = 32; s > 0; s >>= 1) { if (n < s) red[n] = fmaxf(red[n], red[n + s]); __syncthreads(); }
  float m = red[0]; __syncthreads();
  float e = valid ? __expf(sc - m) : 0.f;
  red[n] = e; __syncthreads();
  for (int s = 32; s > 0; s >>= 1) { if (n < s) red[n] += red[n + s]; __syncthreads(); }
  float z = (1.0f - ssw[r]) * e / red[0];
  out2[(long long)r * NE_ + n] = z;
  out1[(long long)r * OUTC_ + TGT_ + n] = logf(z + 1e-6f);
}

// ---------------- host ----------------
extern "C" void kernel_launch(void* const* d_in, const int* in_sizes, int n_in,
                              void* d_out, int out_size, void* d_ws, size_t ws_size,
                              hipStream_t stream) {
  (void)in_sizes; (void)n_in; (void)out_size; (void)ws_size;
  const int*   outp    = (const int*)  d_in[0];
  const float* ents    = (const float*)d_in[1];
  const int*   entlens = (const int*)  d_in[2];
  const float* emb     = (const float*)d_in[3];
  const float* W_ih    = (const float*)d_in[4];
  const float* b_ih    = (const float*)d_in[5];
  const float* W_hh    = (const float*)d_in[6];
  const float* b_hh    = (const float*)d_in[7];
  const float* Wq      = (const float*)d_in[8];
  const float* Wk      = (const float*)d_in[9];
  const float* Wv      = (const float*)d_in[10];
  const float* Wsw     = (const float*)d_in[11];
  const float* bsw     = (const float*)d_in[12];
  const float* Wout    = (const float*)d_in[13];
  const float* bout    = (const float*)d_in[14];
  const float* Wmat    = (const float*)d_in[15];
  const float* bmat    = (const float*)d_in[16];

  float* out1 = (float*)d_out;                               // [2048, 16064] log(full)
  float* out2 = out1 + (long long)ROWS_ * OUTC_;             // [2048, 64]    z

  char* p = (char*)d_ws;
  auto alloc = [&](size_t bytes) -> char* {
    char* r = p; p += (bytes + 255) & ~(size_t)255; return r;
  };
  unsigned short* U_bf    = (unsigned short*)alloc((size_t)H4_ * H2_ * 2);
  float*          bc      = (float*)         alloc((size_t)H4_ * 4);
  unsigned short* We_bf   = (unsigned short*)alloc((size_t)H4_ * H_ * 2);
  unsigned short* e_bf    = (unsigned short*)alloc((size_t)ROWS_ * H_ * 2);
  float*          Ge      = (float*)         alloc((size_t)ROWS_ * H4_ * 4);
  unsigned short* Wq_bf   = (unsigned short*)alloc((size_t)H_ * H_ * 2);
  unsigned short* Wk_bf   = (unsigned short*)alloc((size_t)H_ * H_ * 2);
  unsigned short* Wv_bf   = (unsigned short*)alloc((size_t)H_ * H_ * 2);
  unsigned short* Wmat_bf = (unsigned short*)alloc((size_t)H_ * H2_ * 2);
  unsigned short* Wout_bf = (unsigned short*)alloc((size_t)TGT_ * H2_ * 2);
  unsigned short* ents_bf = (unsigned short*)alloc((size_t)B_ * NE_ * H_ * 2);
  float*          Kbuf    = (float*)         alloc((size_t)B_ * NE_ * H_ * 4);
  float*          Vbuf    = (float*)         alloc((size_t)B_ * NE_ * H_ * 4);
  float*          cxbuf   = (float*)         alloc((size_t)B_ * H_ * 4);
  unsigned short* x_bf    = (unsigned short*)alloc((size_t)B_ * H2_ * 2);
  float*          gbuf    = (float*)         alloc((size_t)B_ * H4_ * 4);
  float*          qbuf    = (float*)         alloc((size_t)B_ * H_ * 4);
  float*          lbuf    = (float*)         alloc((size_t)ROWS_ * H2_ * 4);
  unsigned short* l_bf    = (unsigned short*)alloc((size_t)ROWS_ * H2_ * 2);
  float*          ssw     = (float*)         alloc((size_t)ROWS_ * 4);
  unsigned short* dec_bf  = (unsigned short*)alloc((size_t)ROWS_ * H_ * 2);
  float*          unbuf   = (float*)         alloc((size_t)ROWS_ * NE_ * 4);
  unsigned*       bar     = (unsigned*)      alloc(256);

  auto blocks = [](long long n) { return (unsigned)((n + 255) / 256); };

  // --- prep: bf16 weight copies, fused U, embedding gather, initial state ---
  cvt_f32_bf16<<<blocks((long long)H_ * H_), 256, 0, stream>>>(Wq, Wq_bf, (long long)H_ * H_);
  cvt_f32_bf16<<<blocks((long long)H_ * H_), 256, 0, stream>>>(Wk, Wk_bf, (long long)H_ * H_);
  cvt_f32_bf16<<<blocks((long long)H_ * H_), 256, 0, stream>>>(Wv, Wv_bf, (long long)H_ * H_);
  cvt_f32_bf16<<<blocks((long long)H_ * H2_), 256, 0, stream>>>(Wmat, Wmat_bf, (long long)H_ * H2_);
  cvt_f32_bf16<<<blocks((long long)TGT_ * H2_), 256, 0, stream>>>(Wout, Wout_bf, (long long)TGT_ * H2_);
  cvt_f32_bf16<<<blocks((long long)B_ * NE_ * H_), 256, 0, stream>>>(ents, ents_bf, (long long)B_ * NE_ * H_);
  build_U<<<blocks((long long)H4_ * H2_), 256, 0, stream>>>(W_ih, W_hh, b_ih, b_hh, U_bf, bc);
  build_We<<<blocks((long long)H4_ * H_), 256, 0, stream>>>(W_ih, We_bf);
  gather_emb<<<blocks((long long)ROWS_ * H_), 256, 0, stream>>>(outp, emb, e_bf);
  init_state<<<blocks((long long)B_ * H_), 256, 0, stream>>>(ents, cxbuf, x_bf, bar);

  // --- hoisted GEMMs: Ge = e @ We^T ; K = ents @ Wk^T ; V = ents @ Wv^T ---
  gemm_wmma<<<dim3(8, 32, 1), 256, 0, stream>>>(e_bf, We_bf, Ge, nullptr, nullptr,
      ROWS_, H4_, H_, H_, H_, H4_, 0, 0, 0);
  gemm_wmma<<<dim3(2, 16, 1), 256, 0, stream>>>(ents_bf, Wk_bf, Kbuf, nullptr, nullptr,
      B_ * NE_, H_, H_, H_, H_, H_, 0, 0, 0);
  gemm_wmma<<<dim3(2, 16, 1), 256, 0, stream>>>(ents_bf, Wv_bf, Vbuf, nullptr, nullptr,
      B_ * NE_, H_, H_, H_, H_, H_, 0, 0, 0);

  // --- persistent LSTM + attention recurrence (16 WGPs, phase-barriered) ---
  recurrent_kernel<<<RWG_, 256, 0, stream>>>(x_bf, U_bf, bc, Ge, gbuf, cxbuf, Wq_bf, qbuf,
                                             Kbuf, Vbuf, entlens, lbuf, l_bf, bar);

  // --- epilogues ---
  switch_kernel<<<ROWS_, 256, 0, stream>>>(lbuf, Wsw, bsw, ssw);

  // logits = l @ Wout^T + b_out, written straight into d_out (ldc = 16064)
  gemm_wmma<<<dim3(63, 32, 1), 256, 0, stream>>>(l_bf, Wout_bf, out1, nullptr, bout,
      ROWS_, TGT_, H2_, H2_, H2_, OUTC_, 0, 0, 0);

  // dec = l @ Wmat^T + b_mattn (bf16 only, feeds the un GEMM)
  gemm_wmma<<<dim3(2, 32, 1), 256, 0, stream>>>(l_bf, Wmat_bf, nullptr, dec_bf, bmat,
      ROWS_, H_, H2_, H2_, H2_, H_, 0, 0, 0);

  // un[b] = dec[b] @ ents[b]^T   (batched over b)
  gemm_wmma<<<dim3(1, 2, B_), 256, 0, stream>>>(dec_bf, ents_bf, unbuf, nullptr, nullptr,
      T_, NE_, H_, H_, H_, NE_,
      (long long)T_ * H_, (long long)NE_ * H_, (long long)T_ * NE_);

  softmax_o_kernel<<<ROWS_, 256, 0, stream>>>(out1, ssw);
  z_kernel<<<ROWS_, 64, 0, stream>>>(unbuf, entlens, ssw, out1, out2);
}
